// GIN_HSP_Layer_53609781789206
// MI455X (gfx1250) — matrix operand order repaired and checked by reading, not compiled, not run on hardware
//
#include <hip/hip_runtime.h>
#include <hip/hip_bf16.h>

#define C 128
#define BN_EPS 1e-5f

typedef __attribute__((ext_vector_type(16))) __bf16 v16bf;
typedef __attribute__((ext_vector_type(8)))  float  v8f;
typedef __attribute__((ext_vector_type(4)))  __bf16 bf4;

union Frag {
    uint4 u[2];
    v16bf v;
};

// Guaranteed-native non-returning f32 atomic adds (no CAS-loop legalization):
// 4 contiguous channels share one 64-bit address pair via immediate offsets.
__device__ __forceinline__ void atomic_add_f32x4(float* dst, float a, float b, float c, float d) {
    asm volatile(
        "global_atomic_add_f32 %0, %1, off\n\t"
        "global_atomic_add_f32 %0, %2, off offset:4\n\t"
        "global_atomic_add_f32 %0, %3, off offset:8\n\t"
        "global_atomic_add_f32 %0, %4, off offset:12"
        :
        : "v"(dst), "v"(a), "v"(b), "v"(c), "v"(d)
        : "memory");
}

__device__ __forceinline__ void atomic_add_f32(float* dst, float a) {
    asm volatile("global_atomic_add_f32 %0, %1, off" :: "v"(dst), "v"(a) : "memory");
}

// --- prep: Wt[c_out*C + c_in] = bf16(W[c_in*C + c_out]); zero BN stat accumulators ---
__global__ void prep_kernel(const float* __restrict__ W1, const float* __restrict__ W2,
                            __bf16* __restrict__ Wt1, __bf16* __restrict__ Wt2,
                            float* __restrict__ stats /* 512 floats */) {
    int idx = blockIdx.x * blockDim.x + threadIdx.x;   // 16384 threads
    int co = idx >> 7, ci = idx & (C - 1);
    Wt1[co * C + ci] = (__bf16)W1[ci * C + co];
    Wt2[co * C + ci] = (__bf16)W2[ci * C + co];
    if (idx < 512) stats[idx] = 0.0f;
}

// --- copy: agg = X  (the (1+eps)*X term; EPS_GIN = 0) ---
__global__ void copy_kernel(const float* __restrict__ X, float* __restrict__ A, int total4) {
    int i = blockIdx.x * blockDim.x + threadIdx.x;
    if (i < total4) ((float4*)A)[i] = ((const float4*)X)[i];
}

// --- scatter: one wave per edge; agg[row] += hop_coef[w-1] * X[col] ---
__global__ __launch_bounds__(256)
void scatter_kernel(const float* __restrict__ X,
                    const long long* __restrict__ ei,   // [2,E] flat: rows then cols
                    const long long* __restrict__ ew,   // [E] hop distance 1..K
                    const float* __restrict__ hc,       // [K]
                    float* __restrict__ A, int E) {
    int wave = threadIdx.x >> 5;
    int lane = threadIdx.x & 31;
    int e = blockIdx.x * 8 + wave;
    if (e >= E) return;
    int row = (int)ei[e];
    int col = (int)ei[E + e];
    float coef = hc[(int)ew[e] - 1];
    float4 x = *(const float4*)(X + (size_t)col * C + lane * 4);
    float* dst = A + (size_t)row * C + lane * 4;
    atomic_add_f32x4(dst, coef * x.x, coef * x.y, coef * x.z, coef * x.w);
}

// --- GEMM (act[N,128] @ W[128,128] + bias) via v_wmma_f32_16x16x32_bf16,
//     plus per-channel sum / sumsq accumulation for BatchNorm ---
__global__ __launch_bounds__(128)
void gemm_bn_stats_kernel(const float* __restrict__ act,
                          const __bf16* __restrict__ Wt,   // [c_out][c_in] bf16
                          const float* __restrict__ bias,
                          float* __restrict__ Y,
                          float* __restrict__ sumBuf, float* __restrict__ sqBuf,
                          int nRows) {
    __shared__ __bf16 ldsW[C * C];          // 32 KB, whole transposed weight
    __shared__ __bf16 ldsA[4][16 * C];      // 4 waves x (16 rows x 128 cols) bf16

    int tid = threadIdx.x;
    // cooperative load of Wt into LDS (16-byte vectors)
    {
        const uint4* src = (const uint4*)Wt;
        uint4* dst = (uint4*)ldsW;
        #pragma unroll
        for (int i = 0; i < (C * C) / (8 * 128); ++i)
            dst[tid + i * 128] = src[tid + i * 128];
    }
    __syncthreads();

    int wave = tid >> 5, lane = tid & 31;
    int strip = blockIdx.x * 4 + wave;
    int row0 = strip * 16;
    if (row0 >= nRows) return;              // N % 16 == 0: strips all-or-nothing

    // stage this wave's 16x128 activation strip as bf16 (in-wave DS ops are in order)
    __bf16* la = &ldsA[wave][0];
    #pragma unroll
    for (int r = 0; r < 16; ++r) {
        float4 v = *(const float4*)(act + (size_t)(row0 + r) * C + lane * 4);
        bf4 b;
        b.x = (__bf16)v.x; b.y = (__bf16)v.y; b.z = (__bf16)v.z; b.w = (__bf16)v.w;
        *(bf4*)&la[r * C + lane * 4] = b;
    }

    int m  = lane & 15;      // A row / B column within tile
    int hi = lane >> 4;      // lane half selects K sub-range

    v8f acc[8];
    #pragma unroll
    for (int j = 0; j < 8; ++j) {
        float b = bias[j * 16 + m];
        #pragma unroll
        for (int r = 0; r < 8; ++r) acc[j][r] = b;   // bias pre-seeded into accumulator
    }

    #pragma unroll
    for (int kk = 0; kk < 4; ++kk) {
        // A fragment: lane<16 -> K {0..7, 16..23}; lane>=16 -> K {8..15, 24..31}
        Frag fa;
        const __bf16* pa = &la[m * C + kk * 32 + hi * 8];
        fa.u[0] = *(const uint4*)pa;
        fa.u[1] = *(const uint4*)(pa + 16);
        #pragma unroll
        for (int j = 0; j < 8; ++j) {
            // B fragment: col = j*16+m, lane<16 -> K {0..7, 16..23}; lane>=16 -> K {8..15, 24..31}
            Frag fb;
            const __bf16* pb = &ldsW[(j * 16 + m) * C + kk * 32 + hi * 8];
            fb.u[0] = *(const uint4*)pb;
            fb.u[1] = *(const uint4*)(pb + 16);
            acc[j] = __builtin_amdgcn_wmma_f32_16x16x32_bf16(
                false, fa.v, false, fb.v, (short)0, acc[j], false, false);
        }
    }

    // epilogue: store Y, accumulate per-channel sum / sumsq for BN
    int rowB = row0 + hi * 8;
    #pragma unroll
    for (int j = 0; j < 8; ++j) {
        int col = j * 16 + m;
        float s = 0.f, sq = 0.f;
        #pragma unroll
        for (int r = 0; r < 8; ++r) {
            float y = acc[j][r];
            Y[(size_t)(rowB + r) * C + col] = y;
            s += y; sq += y * y;
        }
        atomic_add_f32(&sumBuf[col], s);
        atomic_add_f32(&sqBuf[col], sq);
    }
}

// --- BatchNorm (training-mode batch stats) + ReLU, element-wise float4 pass ---
__global__ void bn_relu_kernel(const float* __restrict__ Y,
                               const float* __restrict__ sumBuf, const float* __restrict__ sqBuf,
                               const float* __restrict__ g, const float* __restrict__ beta,
                               float* __restrict__ out, int nRows) {
    int i = blockIdx.x * blockDim.x + threadIdx.x;
    int total4 = nRows * (C / 4);
    if (i >= total4) return;
    int c = (i * 4) & (C - 1);
    float invN = 1.0f / (float)nRows;
    float4 y  = ((const float4*)Y)[i];
    float4 s  = *(const float4*)&sumBuf[c];
    float4 q  = *(const float4*)&sqBuf[c];
    float4 gg = *(const float4*)&g[c];
    float4 bb = *(const float4*)&beta[c];

    float m0 = s.x * invN, m1 = s.y * invN, m2 = s.z * invN, m3 = s.w * invN;
    float o0 = fmaxf(0.f, gg.x * (y.x - m0) * rsqrtf(q.x * invN - m0 * m0 + BN_EPS) + bb.x);
    float o1 = fmaxf(0.f, gg.y * (y.y - m1) * rsqrtf(q.y * invN - m1 * m1 + BN_EPS) + bb.y);
    float o2 = fmaxf(0.f, gg.z * (y.z - m2) * rsqrtf(q.z * invN - m2 * m2 + BN_EPS) + bb.z);
    float o3 = fmaxf(0.f, gg.w * (y.w - m3) * rsqrtf(q.w * invN - m3 * m3 + BN_EPS) + bb.w);
    ((float4*)out)[i] = make_float4(o0, o1, o2, o3);
}

extern "C" void kernel_launch(void* const* d_in, const int* in_sizes, int n_in,
                              void* d_out, int out_size, void* d_ws, size_t ws_size,
                              hipStream_t stream) {
    const float*     X   = (const float*)d_in[0];
    const long long* ei  = (const long long*)d_in[1];   // int64 [2,E]
    const long long* ew  = (const long long*)d_in[2];   // int64 [E]
    const float*     W1  = (const float*)d_in[3];
    const float*     b1  = (const float*)d_in[4];
    const float*     g1  = (const float*)d_in[5];
    const float*     be1 = (const float*)d_in[6];
    const float*     W2  = (const float*)d_in[7];
    const float*     b2  = (const float*)d_in[8];
    const float*     g2  = (const float*)d_in[9];
    const float*     be2 = (const float*)d_in[10];
    const float*     hc  = (const float*)d_in[11];

    int N = in_sizes[0] / C;
    int E = in_sizes[2];
    float* out = (float*)d_out;

    // workspace layout
    float*  bufA  = (float*)d_ws;                     // agg, then h  (N*C f32)
    float*  bufY  = bufA + (size_t)N * C;             // pre-BN GEMM out (N*C f32)
    float*  stats = bufY + (size_t)N * C;             // sum1|sq1|sum2|sq2 (4 x 128)
    __bf16* Wt1   = (__bf16*)(stats + 512);           // 128x128 bf16 (transposed)
    __bf16* Wt2   = Wt1 + C * C;

    int total4 = N * (C / 4);
    int strips = N / 16;                              // 3125 (N % 16 == 0)
    int gblk   = (strips + 3) / 4;

    prep_kernel<<<C, C, 0, stream>>>(W1, W2, Wt1, Wt2, stats);
    copy_kernel<<<(total4 + 255) / 256, 256, 0, stream>>>(X, bufA, total4);
    scatter_kernel<<<(E + 7) / 8, 256, 0, stream>>>(X, ei, ew, hc, bufA, E);

    gemm_bn_stats_kernel<<<gblk, 128, 0, stream>>>(bufA, Wt1, b1, bufY, stats, stats + 128, N);
    bn_relu_kernel<<<(total4 + 255) / 256, 256, 0, stream>>>(bufY, stats, stats + 128, g1, be1, bufA, N);

    gemm_bn_stats_kernel<<<gblk, 128, 0, stream>>>(bufA, Wt2, b2, bufY, stats + 256, stats + 384, N);
    bn_relu_kernel<<<(total4 + 255) / 256, 256, 0, stream>>>(bufY, stats + 256, stats + 384, g2, be2, out, N);
}
